// TransformerBlock_48275432407846
// MI455X (gfx1250) — matrix operand order repaired and checked by reading, not compiled, not run on hardware
//
#include <hip/hip_runtime.h>

typedef __attribute__((ext_vector_type(16))) __bf16 v16bf;
typedef __attribute__((ext_vector_type(8)))  __bf16 v8bf;
typedef __attribute__((ext_vector_type(4)))  __bf16 v4bf;
typedef __attribute__((ext_vector_type(2)))  __bf16 v2bf;
typedef __attribute__((ext_vector_type(8)))  float  v8f;

constexpr int kB  = 2;
constexpr int kS  = 1024;
constexpr int kD  = 1024;
constexpr int kH  = 16;
constexpr int kHD = 64;
constexpr int kF  = 4096;
constexpr int kE  = 8;
constexpr int kTOK = kB * kS;   // 2048

// ---------- bf16 conversion: native fptrunc (RNE on gfx1250 bf16 HW) ----------
__device__ __forceinline__ __bf16 f2bf(float f) { return (__bf16)f; }

__device__ __forceinline__ v2bf f2bf2(float a, float b) {
  v2bf r; r.x = (__bf16)a; r.y = (__bf16)b; return r;
}

// convert float4 -> 4 x bf16 and store as one 8-byte LDS store
__device__ __forceinline__ void store_bf4(__bf16* dst, float4 f) {
  v4bf o;
  o.x = (__bf16)f.x; o.y = (__bf16)f.y; o.z = (__bf16)f.z; o.w = (__bf16)f.w;
  *(v4bf*)dst = o;
}

__device__ __forceinline__ float gelu_exact(float x) {
  return 0.5f * x * (1.0f + erff(x * 0.70710678118654752f));
}

// ---------- cross-lane xor-shuffle without LDS round-trips ----------
// masks 1/2/4 stay inside an 8-lane group -> v_mov_b32 dpp8;
// mask 8 stays inside a 16-lane row     -> v_permlane16_b32.
template<int M>
__device__ __forceinline__ float fshfl_xor(float x) {
#if __has_builtin(__builtin_amdgcn_mov_dpp8) && __has_builtin(__builtin_amdgcn_permlane16)
  int v = __builtin_bit_cast(int, x);
  if constexpr (M < 8) {
    constexpr unsigned imm =
        (unsigned)(((0 ^ M) & 7)       | (((1 ^ M) & 7) << 3)  |
                   (((2 ^ M) & 7) << 6)| (((3 ^ M) & 7) << 9)  |
                   (((4 ^ M) & 7) <<12)| (((5 ^ M) & 7) << 15) |
                   (((6 ^ M) & 7) <<18)| (((7 ^ M) & 7) << 21));
    v = __builtin_amdgcn_mov_dpp8(v, imm);
  } else {
    // lane i <- lane i^8 within each 16-lane row
    v = __builtin_amdgcn_permlane16(v, v, 0xFEDCBA98u, 0x76543210u, false, false);
  }
  return __builtin_bit_cast(float, v);
#else
  return __shfl_xor(x, M);
#endif
}

// async 16B global->LDS copy (ASYNCcnt-tracked, bypasses VGPRs).
// LDS byte address = low 32 bits of the flat shared-pointer (ISA: LDS aperture
// uses addr[31:0] as the wave-relative LDS address).
__device__ __forceinline__ void async_copy16(void* lds_dst, const void* gsrc) {
  unsigned           ldsoff = (unsigned)(size_t)lds_dst;
  unsigned long long gaddr  = (unsigned long long)(size_t)gsrc;
  asm volatile("global_load_async_to_lds_b128 %0, %1, off"
               :: "v"(ldsoff), "v"(gaddr) : "memory");
}
__device__ __forceinline__ void async_wait0() {
  asm volatile("s_wait_asynccnt 0x0" ::: "memory");
}

// A-fragment (16x32 bf16): lane m=lane&15, K-base (lane>>4)*8, quads at +0 and +16
__device__ __forceinline__ v16bf load_fragA(const __bf16* t, int ld, int kb0) {
  const int lane = threadIdx.x & 31;
  const __bf16* p = t + (lane & 15) * ld + (((lane >> 4) << 3) + kb0);
  v8bf lo = *(const v8bf*)p;
  v8bf hi = *(const v8bf*)(p + 16);
  return __builtin_shufflevector(lo, hi, 0,1,2,3,4,5,6,7,8,9,10,11,12,13,14,15);
}

// B-fragment (32x16 bf16) from B^T stored row-major [n][k]: lane n=lane&15,
// K-base (lane>>4)*8, quads at +0 and +16.
__device__ __forceinline__ v16bf load_fragB(const __bf16* t, int ld, int kb0) {
  const int lane = threadIdx.x & 31;
  const __bf16* p = t + (lane & 15) * ld + (((lane >> 4) << 3) + kb0);
  v8bf lo = *(const v8bf*)p;
  v8bf hi = *(const v8bf*)(p + 16);
  return __builtin_shufflevector(lo, hi, 0,1,2,3,4,5,6,7,8,9,10,11,12,13,14,15);
}

__device__ __forceinline__ v8f wmma_bf16(v16bf a, v16bf b, v8f c) {
  return __builtin_amdgcn_wmma_f32_16x16x32_bf16(false, a, false, b, (short)0, c, false, false);
}

// ---------- generic WMMA GEMM, 128x128x32 tiles, 256 threads (8 waves) ----------
// MODE 0: C[M,N] = A_f32[M,K] @ B[K,N] (+bias)
// MODE 1: h[offs[e]+m, N] = bf16(gelu(gather(A,list)[m] @ B_e + bias_e))   (rows < cnt[e])
// MODE 2: atomicAdd(C[list[m], N], gate[m] * (Abf[offs[e]+m] @ B_e + bias_e))
template<int MODE>
__global__ void __launch_bounds__(256)
gemm_wmma_kernel(const float* __restrict__ A, const __bf16* __restrict__ Abf,
                 const float* __restrict__ Bm, const float* __restrict__ bias,
                 float* __restrict__ Cout, __bf16* __restrict__ Hout,
                 const int* __restrict__ list, const float* __restrict__ gates,
                 const int* __restrict__ cnt, const int* __restrict__ offs,
                 int M, int N, int K)
{
  __shared__ __align__(16) __bf16 As[128][48];
  __shared__ __align__(16) __bf16 Bs[128][48];

  const int tid  = threadIdx.x;
  const int wave = tid >> 5;
  const int lane = tid & 31;
  const int nt = blockIdx.x, mt = blockIdx.y, e = blockIdx.z;

  int mcount = M;
  int mbase  = mt * 128;
  const int*   lst = nullptr;
  const float* gts = nullptr;
  const float* bias_e = bias;
  const float* Bp = Bm;
  int hbase = 0;
  if constexpr (MODE != 0) {
    mcount = cnt[e];
    if (mbase >= mcount) return;
    lst = list + e * kTOK;
    gts = gates + e * kTOK;
    Bp  = Bm + (size_t)e * K * N;
    bias_e = bias + (size_t)e * N;
    hbase = offs[e];
  }

  v8f acc[2][4];
  #pragma unroll
  for (int i = 0; i < 2; ++i)
    #pragma unroll
    for (int j = 0; j < 4; ++j) acc[i][j] = (v8f){};

  const int wr = wave >> 1;   // 0..3 : 32-row group
  const int wc = wave & 1;    // 0..1 : 64-col group

  for (int k0 = 0; k0 < K; k0 += 32) {
    __syncthreads();
    // ---- stage A tile (128 x 32) as bf16 ----
    if constexpr (MODE == 2) {
      // h is already bf16 in global: pure copy -> async global->LDS (B128)
      #pragma unroll
      for (int i = tid; i < 512; i += 256) {            // 128 rows x 4 chunks of 8
        int r = i >> 2, c8 = (i & 3) << 3;
        int ml = min(mbase + r, mcount - 1);
        const __bf16* src = Abf + ((size_t)(hbase + ml)) * K + k0 + c8;
        async_copy16(&As[r][c8], src);
      }
    } else {
      #pragma unroll
      for (int i = tid; i < 1024; i += 256) {           // 128 rows x 8 float4
        int r = i >> 3, c4 = (i & 7) << 2;
        int grow;
        if constexpr (MODE == 0) grow = mbase + r;
        else                     grow = lst[min(mbase + r, mcount - 1)];
        float4 f = *(const float4*)(A + (size_t)grow * K + k0 + c4);
        store_bf4(&As[r][c4], f);
      }
    }
    // ---- stage B tile (32 x 128) transposed -> Bs[n][k] bf16 ----
    #pragma unroll
    for (int i = tid; i < 1024; i += 256) {             // 32 rows x 32 float4
      int kk = i >> 5, c4 = (i & 31) << 2;
      float4 f = *(const float4*)(Bp + (size_t)(k0 + kk) * N + nt * 128 + c4);
      Bs[c4 + 0][kk] = f2bf(f.x);
      Bs[c4 + 1][kk] = f2bf(f.y);
      Bs[c4 + 2][kk] = f2bf(f.z);
      Bs[c4 + 3][kk] = f2bf(f.w);
    }
    if constexpr (MODE == 2) async_wait0();
    __syncthreads();

    v16bf fa0 = load_fragA(&As[wr * 32][0],      48, 0);
    v16bf fa1 = load_fragA(&As[wr * 32 + 16][0], 48, 0);
    #pragma unroll
    for (int j = 0; j < 4; ++j) {
      v16bf fb = load_fragB(&Bs[wc * 64 + j * 16][0], 48, 0);
      acc[0][j] = wmma_bf16(fa0, fb, acc[0][j]);
      acc[1][j] = wmma_bf16(fa1, fb, acc[1][j]);
    }
  }

  // ---- epilogue ----
  #pragma unroll
  for (int i = 0; i < 2; ++i) {
    #pragma unroll
    for (int j = 0; j < 4; ++j) {
      const int col = nt * 128 + wc * 64 + j * 16 + (lane & 15);
      #pragma unroll
      for (int r = 0; r < 8; ++r) {
        const int ml = wr * 32 + i * 16 + ((lane >> 4) << 3) + r;  // row in tile
        const int mg = mbase + ml;
        float v = acc[i][j][r];
        if constexpr (MODE == 0) {
          if (bias) v += bias[col];
          Cout[(size_t)mg * N + col] = v;
        } else if constexpr (MODE == 1) {
          if (mg < mcount) {
            v = gelu_exact(v + bias_e[col]);
            Hout[((size_t)(hbase + mg)) * N + col] = f2bf(v);
          }
        } else {
          if (mg < mcount) {
            int tok = lst[mg];
            float g = gts[mg];
            atomicAdd(&Cout[(size_t)tok * N + col], g * (v + bias_e[col]));
          }
        }
      }
    }
  }
}

// ---------- flash attention: grid(S/64, B*H), 128 threads (4 waves x 16 q-rows) ----------
__global__ void __launch_bounds__(128)
flash_attn_kernel(const float* __restrict__ Q, const float* __restrict__ Kb,
                  const float* __restrict__ Vb, float* __restrict__ Ctx)
{
  __shared__ __align__(16) __bf16 q_lds[64][80];
  __shared__ __align__(16) __bf16 k_lds[64][80];   // [key][hd]
  __shared__ __align__(16) __bf16 v_lds[64][80];   // [hd][key]  (transposed)
  __shared__ __align__(16) __bf16 p_lds[64][80];

  const int tid  = threadIdx.x;
  const int wave = tid >> 5;
  const int lane = tid & 31;
  const int qt = blockIdx.x;
  const int bh = blockIdx.y;
  const int b = bh >> 4, h = bh & 15;

  const float* qp = Q   + ((size_t)b * kS) * kD + h * kHD;
  const float* kp = Kb  + ((size_t)b * kS) * kD + h * kHD;
  const float* vp = Vb  + ((size_t)b * kS) * kD + h * kHD;
  float*       op = Ctx + ((size_t)b * kS) * kD + h * kHD;

  // load 64x64 Q tile once; fold the 1/sqrt(HD)=0.125 score scale into Q here
  for (int i = tid; i < 64 * 16; i += 128) {
    int r = i >> 4, c4 = (i & 15) << 2;
    float4 f = *(const float4*)(qp + (size_t)(qt * 64 + r) * kD + c4);
    f.x *= 0.125f; f.y *= 0.125f; f.z *= 0.125f; f.w *= 0.125f;
    store_bf4(&q_lds[r][c4], f);
  }

  v8f ctx[4];
  #pragma unroll
  for (int j = 0; j < 4; ++j) ctx[j] = (v8f){};
  float mrun[8], lrun[8];
  #pragma unroll
  for (int r = 0; r < 8; ++r) { mrun[r] = -1e30f; lrun[r] = 0.0f; }

  const int m0 = wave * 16;

  for (int kt = 0; kt < kS / 64; ++kt) {
    __syncthreads();
    // stage K (row-major) and V (transposed) tiles
    for (int i = tid; i < 64 * 16; i += 128) {
      int r = i >> 4, c4 = (i & 15) << 2;
      float4 f = *(const float4*)(kp + (size_t)(kt * 64 + r) * kD + c4);
      store_bf4(&k_lds[r][c4], f);
      float4 g = *(const float4*)(vp + (size_t)(kt * 64 + r) * kD + c4);
      v_lds[c4 + 0][r] = f2bf(g.x);
      v_lds[c4 + 1][r] = f2bf(g.y);
      v_lds[c4 + 2][r] = f2bf(g.z);
      v_lds[c4 + 3][r] = f2bf(g.w);
    }
    __syncthreads();

    // scores S = (Q/8) K^T   (scale pre-folded into Q)
    v8f s[4];
    #pragma unroll
    for (int j = 0; j < 4; ++j) s[j] = (v8f){};
    #pragma unroll
    for (int kk = 0; kk < 64; kk += 32) {
      v16bf fa = load_fragA(&q_lds[m0][0], 80, kk);
      #pragma unroll
      for (int j = 0; j < 4; ++j) {
        v16bf fb = load_fragB(&k_lds[j * 16][0], 80, kk);
        s[j] = wmma_bf16(fa, fb, s[j]);
      }
    }

    // online softmax: rows r live in VGPR r across the 16 lanes of the half-wave
    #pragma unroll
    for (int r = 0; r < 8; ++r) {
      float tm = -1e30f;
      #pragma unroll
      for (int j = 0; j < 4; ++j) tm = fmaxf(tm, s[j][r]);
      tm = fmaxf(tm, fshfl_xor<1>(tm));
      tm = fmaxf(tm, fshfl_xor<2>(tm));
      tm = fmaxf(tm, fshfl_xor<4>(tm));
      tm = fmaxf(tm, fshfl_xor<8>(tm));
      float mnew = fmaxf(mrun[r], tm);
      float corr = __expf(mrun[r] - mnew);
      float ts = 0.0f;
      #pragma unroll
      for (int j = 0; j < 4; ++j) {
        float p = __expf(s[j][r] - mnew);
        s[j][r] = p;
        ts += p;
      }
      ts += fshfl_xor<1>(ts);
      ts += fshfl_xor<2>(ts);
      ts += fshfl_xor<4>(ts);
      ts += fshfl_xor<8>(ts);
      lrun[r] = lrun[r] * corr + ts;
      mrun[r] = mnew;
      #pragma unroll
      for (int j = 0; j < 4; ++j) ctx[j][r] *= corr;
    }

    // spill P to (wave-private rows of) LDS in bf16 to re-read as A-fragments
    #pragma unroll
    for (int j = 0; j < 4; ++j)
      #pragma unroll
      for (int r = 0; r < 8; ++r)
        p_lds[m0 + ((lane >> 4) << 3) + r][j * 16 + (lane & 15)] = f2bf(s[j][r]);
    asm volatile("s_wait_dscnt 0x0" ::: "memory");

    // ctx += P @ V
    #pragma unroll
    for (int kk = 0; kk < 64; kk += 32) {
      v16bf fa = load_fragA(&p_lds[m0][0], 80, kk);
      #pragma unroll
      for (int j = 0; j < 4; ++j) {
        v16bf fb = load_fragB(&v_lds[j * 16][0], 80, kk);
        ctx[j] = wmma_bf16(fa, fb, ctx[j]);
      }
    }
  }

  #pragma unroll
  for (int j = 0; j < 4; ++j)
    #pragma unroll
    for (int r = 0; r < 8; ++r) {
      int row = qt * 64 + m0 + ((lane >> 4) << 3) + r;
      int col = j * 16 + (lane & 15);
      op[(size_t)row * kD + col] = ctx[j][r] / lrun[r];
    }
}

// ---------- fused add + LayerNorm: one 256-thread block per row ----------
__global__ void __launch_bounds__(256)
add_ln_kernel(const float* __restrict__ X, const float* __restrict__ Y,
              const float* __restrict__ g, const float* __restrict__ bta,
              float* __restrict__ out)
{
  __shared__ float s1[256], s2[256];
  const int row = blockIdx.x, tid = threadIdx.x;
  float vals[4];
  float sum = 0.0f, sq = 0.0f;
  #pragma unroll
  for (int i = 0; i < 4; ++i) {
    int d = tid + i * 256;
    float v = X[(size_t)row * kD + d] + Y[(size_t)row * kD + d];
    vals[i] = v; sum += v; sq += v * v;
  }
  s1[tid] = sum; s2[tid] = sq;
  __syncthreads();
  for (int s = 128; s > 0; s >>= 1) {
    if (tid < s) { s1[tid] += s1[tid + s]; s2[tid] += s2[tid + s]; }
    __syncthreads();
  }
  float mean = s1[0] * (1.0f / kD);
  float var  = s2[0] * (1.0f / kD) - mean * mean;
  float inv  = rsqrtf(var + 1e-6f);
  #pragma unroll
  for (int i = 0; i < 4; ++i) {
    int d = tid + i * 256;
    out[(size_t)row * kD + d] = (vals[i] - mean) * inv * g[d] + bta[d];
  }
}

// ---------- router: softmax over E=8 logits, top-2 + renormalize ----------
__global__ void __launch_bounds__(256)
router_kernel(const float* __restrict__ out1, const float* __restrict__ Wr,
              int* __restrict__ top_i, float* __restrict__ top_g)
{
  __shared__ float red[256][kE];
  const int t = blockIdx.x, tid = threadIdx.x;
  float a[kE];
  #pragma unroll
  for (int e = 0; e < kE; ++e) a[e] = 0.0f;
  for (int d = tid; d < kD; d += 256) {
    float x = out1[(size_t)t * kD + d];
    const float* w = Wr + (size_t)d * kE;
    #pragma unroll
    for (int e = 0; e < kE; ++e) a[e] += x * w[e];
  }
  #pragma unroll
  for (int e = 0; e < kE; ++e) red[tid][e] = a[e];
  __syncthreads();
  for (int s = 128; s > 0; s >>= 1) {
    if (tid < s)
      #pragma unroll
      for (int e = 0; e < kE; ++e) red[tid][e] += red[tid + s][e];
    __syncthreads();
  }
  if (tid == 0) {
    float lg[kE];
    float mx = -1e30f;
    #pragma unroll
    for (int e = 0; e < kE; ++e) mx = fmaxf(mx, red[0][e]);
    float sum = 0.0f;
    #pragma unroll
    for (int e = 0; e < kE; ++e) { lg[e] = __expf(red[0][e] - mx); sum += lg[e]; }
    #pragma unroll
    for (int e = 0; e < kE; ++e) lg[e] /= sum;
    int i0 = 0;
    for (int e = 1; e < kE; ++e) if (lg[e] > lg[i0]) i0 = e;
    int i1 = (i0 == 0) ? 1 : 0;
    for (int e = 0; e < kE; ++e) if (e != i0 && lg[e] > lg[i1]) i1 = e;
    float p0 = lg[i0], p1 = lg[i1];
    float e1 = __expf(p1 - p0);          // softmax over [p0,p1]
    float gs = 1.0f + e1;
    top_i[t * 2 + 0] = i0; top_i[t * 2 + 1] = i1;
    top_g[t * 2 + 0] = 1.0f / gs; top_g[t * 2 + 1] = e1 / gs;
  }
}

// ---------- deterministic per-expert token lists + compact offsets ----------
__global__ void build_lists_kernel(const int* __restrict__ top_i, const float* __restrict__ top_g,
                                   int* __restrict__ list, float* __restrict__ gates,
                                   int* __restrict__ cnt, int* __restrict__ offs)
{
  const int e = threadIdx.x;
  if (e < kE) {
    int c = 0;
    for (int t = 0; t < kTOK; ++t)
      for (int s2 = 0; s2 < 2; ++s2)
        if (top_i[t * 2 + s2] == e) {
          list[e * kTOK + c]  = t;
          gates[e * kTOK + c] = top_g[t * 2 + s2];
          ++c;
        }
    cnt[e] = c;
  }
  __syncthreads();
  if (threadIdx.x == 0) {
    int o = 0;
    for (int e2 = 0; e2 < kE; ++e2) { offs[e2] = o; o += cnt[e2]; }
  }
}

__global__ void zero_kernel(float* __restrict__ p, int n) {
  int i = blockIdx.x * 256 + threadIdx.x;
  if (i < n) p[i] = 0.0f;
}

// ---------- host launcher ----------
extern "C" void kernel_launch(void* const* d_in, const int* in_sizes, int n_in,
                              void* d_out, int out_size, void* d_ws, size_t ws_size,
                              hipStream_t stream) {
  const float* x     = (const float*)d_in[0];
  const float* Wq    = (const float*)d_in[1];
  const float* Wk    = (const float*)d_in[2];
  const float* Wv    = (const float*)d_in[3];
  const float* Wo    = (const float*)d_in[4];
  const float* bo    = (const float*)d_in[5];
  const float* Wr    = (const float*)d_in[6];
  const float* We1   = (const float*)d_in[7];
  const float* be1   = (const float*)d_in[8];
  const float* We2   = (const float*)d_in[9];
  const float* be2   = (const float*)d_in[10];
  const float* ln1_g = (const float*)d_in[11];
  const float* ln1_b = (const float*)d_in[12];
  const float* ln2_g = (const float*)d_in[13];
  const float* ln2_b = (const float*)d_in[14];

  char* ws = (char*)d_ws;
  const size_t MB = 1u << 20;
  float*  Qb   = (float*)(ws + 0 * MB);
  float*  Kb   = (float*)(ws + 8 * MB);
  float*  Vb   = (float*)(ws + 16 * MB);
  float*  CTX  = (float*)(ws + 24 * MB);
  float*  ATT  = (float*)(ws + 32 * MB);
  float*  OUT1 = (float*)(ws + 40 * MB);
  float*  ACC  = (float*)(ws + 48 * MB);
  __bf16* Hb   = (__bf16*)(ws + 56 * MB);          // 2*kTOK rows x kF bf16 = 32MB
  size_t  mo   = 88 * MB;
  int*    TOPI = (int*)  (ws + mo);
  float*  TOPG = (float*)(ws + mo + (16u << 10));
  int*    LIST = (int*)  (ws + mo + (32u << 10));
  float*  GATE = (float*)(ws + mo + (96u << 10));
  int*    CNT  = (int*)  (ws + mo + (160u << 10));
  int*    OFFS = CNT + 16;

  dim3 blk(256);
  dim3 gProj(kD / 128, kTOK / 128, 1);   // 8 x 16

  // QKV projections
  gemm_wmma_kernel<0><<<gProj, blk, 0, stream>>>(x, nullptr, Wq, nullptr, Qb, nullptr,
      nullptr, nullptr, nullptr, nullptr, kTOK, kD, kD);
  gemm_wmma_kernel<0><<<gProj, blk, 0, stream>>>(x, nullptr, Wk, nullptr, Kb, nullptr,
      nullptr, nullptr, nullptr, nullptr, kTOK, kD, kD);
  gemm_wmma_kernel<0><<<gProj, blk, 0, stream>>>(x, nullptr, Wv, nullptr, Vb, nullptr,
      nullptr, nullptr, nullptr, nullptr, kTOK, kD, kD);

  // attention
  flash_attn_kernel<<<dim3(kS / 64, kB * kH), dim3(128), 0, stream>>>(Qb, Kb, Vb, CTX);

  // output projection + bias, then residual+LN1
  gemm_wmma_kernel<0><<<gProj, blk, 0, stream>>>(CTX, nullptr, Wo, bo, ATT, nullptr,
      nullptr, nullptr, nullptr, nullptr, kTOK, kD, kD);
  add_ln_kernel<<<kTOK, blk, 0, stream>>>(x, ATT, ln1_g, ln1_b, OUT1);

  // MoE
  zero_kernel<<<(kTOK * kD + 255) / 256, blk, 0, stream>>>(ACC, kTOK * kD);
  router_kernel<<<kTOK, blk, 0, stream>>>(OUT1, Wr, TOPI, TOPG);
  build_lists_kernel<<<1, 32, 0, stream>>>(TOPI, TOPG, LIST, GATE, CNT, OFFS);

  gemm_wmma_kernel<1><<<dim3(kF / 128, kTOK / 128, kE), blk, 0, stream>>>(
      OUT1, nullptr, We1, be1, nullptr, Hb, LIST, GATE, CNT, OFFS, kTOK, kF, kD);
  gemm_wmma_kernel<2><<<dim3(kD / 128, kTOK / 128, kE), blk, 0, stream>>>(
      nullptr, Hb, We2, be2, ACC, nullptr, LIST, GATE, CNT, OFFS, kTOK, kD, kF);

  // residual + LN2 -> output
  add_ln_kernel<<<kTOK, blk, 0, stream>>>(OUT1, ACC, ln2_g, ln2_b, (float*)d_out);
}
